// LoRALinear1d_2310692405890
// MI455X (gfx1250) — matrix operand order repaired and checked by reading, not compiled, not run on hardware
//
#include <hip/hip_runtime.h>
#include <hip/hip_bf16.h>
#include <stdint.h>

// Problem constants (match reference)
constexpr int kB    = 8;
constexpr int kCIN  = 512;
constexpr int kCOUT = 512;
constexpr int kCINFO= 256;
constexpr int kR    = 8;
constexpr int kL    = 4096;

typedef __bf16 v16bf __attribute__((ext_vector_type(16)));
typedef float  v8f   __attribute__((ext_vector_type(8)));
typedef unsigned int v4u  __attribute__((ext_vector_type(4)));
typedef int          v8i_ __attribute__((ext_vector_type(8)));
typedef int          v4i_ __attribute__((ext_vector_type(4)));

union FragU { uint4 q[2]; v16bf v; };   // 32B fragment assembled from 2x b128

// ---------------------------------------------------------------------------
// Kernel 1: weight-norm main weight -> bf16 row-major (COUT x CIN).
// ---------------------------------------------------------------------------
__global__ void prep_main_w(const float* __restrict__ main_v,
                            const float* __restrict__ main_g,
                            __bf16* __restrict__ Wbf) {
  const int wave = threadIdx.x >> 5, lane = threadIdx.x & 31;
  const int row = blockIdx.x * 8 + wave;              // 512 rows total
  const float* vr = main_v + (size_t)row * kCIN;
  float s = 0.f;
  #pragma unroll
  for (int i = 0; i < kCIN / 32; ++i) { float v = vr[lane + 32 * i]; s += v * v; }
  #pragma unroll
  for (int off = 16; off; off >>= 1) s += __shfl_xor(s, off, 32);
  const float scale = main_g[row] * rsqrtf(s);        // g / ||v||
  #pragma unroll
  for (int i = 0; i < kCIN / 32; ++i)
    Wbf[(size_t)row * kCIN + lane + 32 * i] = (__bf16)(vr[lane + 32 * i] * scale);
}

// ---------------------------------------------------------------------------
// Kernel 2: adapters. a_in -> bf16 [B][16][CIN] (rows 8..15 zeroed elsewhere),
// a_out -> f32 [B][R][COUT]. One wave32 per adapter row.
// ---------------------------------------------------------------------------
__global__ void prep_adapters(const float* __restrict__ g,
                              const float* __restrict__ ain_v,
                              const float* __restrict__ ain_g,
                              const float* __restrict__ ain_b,
                              const float* __restrict__ aout_v,
                              const float* __restrict__ aout_g,
                              const float* __restrict__ aout_b,
                              __bf16* __restrict__ ainT,
                              float*  __restrict__ aoutW) {
  const int wave = threadIdx.x >> 5, lane = threadIdx.x & 31;
  const int row = blockIdx.x * 8 + wave;              // 0..8191
  const bool isIn = row < kCIN * kR;
  const int  row2 = isIn ? row : row - kCIN * kR;
  const float* vr = (isIn ? ain_v : aout_v) + (size_t)row2 * kCINFO;

  float vv[kCINFO / 32];
  float s = 0.f;
  #pragma unroll
  for (int i = 0; i < kCINFO / 32; ++i) { vv[i] = vr[lane + 32 * i]; s += vv[i] * vv[i]; }
  #pragma unroll
  for (int off = 16; off; off >>= 1) s += __shfl_xor(s, off, 32);

  const float gg   = isIn ? ain_g[row2] : aout_g[row2];
  const float bias = isIn ? ain_b[row2] : aout_b[row2];
  const float scale = gg * rsqrtf(s);

  for (int b = 0; b < kB; ++b) {
    const float* gv = g + (size_t)b * kCINFO;         // g is (B, CINFO, 1)
    float d = 0.f;
    #pragma unroll
    for (int i = 0; i < kCINFO / 32; ++i) d += vv[i] * gv[lane + 32 * i];
    #pragma unroll
    for (int off = 16; off; off >>= 1) d += __shfl_xor(d, off, 32);
    const float res = d * scale + bias;
    if (lane == 0) {
      if (isIn) {        // col j -> (cin=j/R, r=j%R); store transposed [b][r][cin]
        const int cin = row2 / kR, r = row2 % kR;
        ainT[((size_t)b * 16 + r) * kCIN + cin] = (__bf16)res;
      } else {           // col j -> (r=j/COUT, o=j%COUT)
        const int r = row2 / kCOUT, o = row2 % kCOUT;
        aoutW[((size_t)b * kR + r) * kCOUT + o] = res;
      }
    }
  }
}

// Zero rows 8..15 of the padded a_in tensor (WMMA M-pad contributes 0).
__global__ void zero_pad_ain(__bf16* __restrict__ ainT) {
  const int idx = blockIdx.x * blockDim.x + threadIdx.x;   // < 8*8*512
  const int b = idx / (8 * kCIN);
  const int rem = idx % (8 * kCIN);
  const int r = rem / kCIN, cin = rem % kCIN;
  ainT[((size_t)b * 16 + 8 + r) * kCIN + cin] = (__bf16)0.0f;
}

// ---------------------------------------------------------------------------
// Kernel 3: fused GEMM  out = W@x + b + a_out^T (a_in^T @ x)
// Block = 256 threads = 8 wave32s (2x4), tile 128(M) x 128(N), K-step 32.
// W tile staged by the Tensor Data Mover (TDM pads LDS rows to 56 elems);
// x tile staged column-wise so B-fragments are 2x ds_load_b128 per lane.
// ---------------------------------------------------------------------------
__launch_bounds__(256)
__global__ void lora_gemm(const float*  __restrict__ x,
                          const __bf16* __restrict__ Wbf,
                          const float*  __restrict__ main_b,
                          const __bf16* __restrict__ ainT,
                          const float*  __restrict__ aoutW,
                          float* __restrict__ out) {
  // Row strides chosen so every lane's two 16B fragment runs are 16B-aligned
  // and the dword stride (28 / 28 / 260) spreads 16 lanes across banks.
  __shared__ __align__(16) __bf16 As [128][56];  // [m][k], TDM-padded rows
  __shared__ __align__(16) __bf16 Xsf[128][56];  // [n][k]  (B-fragment order)
  __shared__ __align__(16) __bf16 Ain[16][520];  // [r][k] full-K a_in slab
  __shared__ float Tld[16][132];                 // t tile (rank rows x N)
  __shared__ float Bld[128];                     // bias tile
  __shared__ float Aout[8][128];                 // a_out tile [r][m]

  const int tid  = threadIdx.x;
  const int lane = tid & 31, wave = tid >> 5;
  const int wm = wave >> 2, wn = wave & 3;       // 2 x 4 wave grid
  const int hl = lane >> 4, l16 = lane & 15;     // lane half / low-16 id
  const int n0 = blockIdx.x * 128, m0 = blockIdx.y * 128, b = blockIdx.z;

  // x staging assignment: column xn, k-half xkh (16 rows per thread)
  const int xn  = tid & 127;
  const int xkh = (tid >> 7) * 16;

  // LDS byte offset of As (flat LDS addresses have offset in low 32 bits)
  const unsigned ldsAs = (unsigned)(uintptr_t)&As[0][0];

  // Prologue: batch-constant LDS tiles
  for (int i = tid; i < 16 * kCIN; i += 256)
    Ain[i >> 9][i & 511] = ainT[(size_t)b * 16 * kCIN + i];
  if (tid < 128) Bld[tid] = main_b[m0 + tid];
  for (int i = tid; i < 8 * 128; i += 256)
    Aout[i >> 7][i & 127] = aoutW[((size_t)b * kR + (i >> 7)) * kCOUT + m0 + (i & 127)];

  const v8f vz = {0.f, 0.f, 0.f, 0.f, 0.f, 0.f, 0.f, 0.f};
  v8f acc[4][2];
  #pragma unroll
  for (int i = 0; i < 4; ++i) { acc[i][0] = vz; acc[i][1] = vz; }
  v8f tacc0 = vz, tacc1 = vz;

  // TDM descriptor group1 (kk-invariant): data_size=2B, pad_enable,
  // pad_interval=16 dwords (one 32-elem row), pad_amount=12 dwords (->56 elem
  // stride), tensor_dim0=512, tensor_dim1=128, tile=32x128, dim0_stride=512.
  v8i_ g1;
  g1[0] = 0x16D10000;      // pad_amount=11(12dw)<<25 | interval=3<<22 | pad_en | data_size=1<<16
  g1[1] = (512 << 16);     // tensor_dim0[15:0] at bits 63:48
  g1[2] = (128 << 16);     // tensor_dim1[15:0] at bits 95:80
  g1[3] = (32  << 16);     // tile_dim0 at bits 127:112
  g1[4] = 128;             // tile_dim1
  g1[5] = 512;             // tensor_dim0_stride[31:0]
  g1[6] = 0; g1[7] = 0;
  const v4i_ z4 = {0, 0, 0, 0};
  const v8i_ z8 = {0, 0, 0, 0, 0, 0, 0, 0};

  __syncthreads();

  for (int kk = 0; kk < kCIN; kk += 32) {
    // --- Stage A (W tile, bf16, no conversion) via Tensor Data Mover -------
    if (wave == 0) {
      const uint64_t ga = (uint64_t)(uintptr_t)(Wbf + (size_t)m0 * kCIN + kk);
      v4u g0;
      g0.x = 1u;                                         // count=1 (valid user D#)
      g0.y = ldsAs;                                      // lds_addr
      g0.z = (unsigned)(ga & 0xFFFFFFFFu);               // global_addr[31:0]
      g0.w = (unsigned)((ga >> 32) & 0x1FFFFFFu) | 0x80000000u; // [56:32] | type=2
      __builtin_amdgcn_tensor_load_to_lds(g0, g1, z4, z4, z8, 0);
    }
    // --- Stage X: 32x128 f32 -> bf16, written in B-fragment order ----------
    {
      union { __bf16 h[16]; uint4 q[2]; } xb;
      const float* xcol = x + ((size_t)b * kCIN + kk + xkh) * kL + n0 + xn;
      #pragma unroll
      for (int j = 0; j < 16; ++j) xb.h[j] = (__bf16)xcol[(size_t)j * kL];
      *(uint4*)&Xsf[xn][xkh]     = xb.q[0];
      *(uint4*)&Xsf[xn][xkh + 8] = xb.q[1];
    }
    if (kk + 32 < kCIN && lane == 0)   // prefetch next x K-slab
      __builtin_prefetch(x + ((size_t)b * kCIN + kk + 32 + (wave << 2)) * kL + n0, 0, 0);

    if (wave == 0) __builtin_amdgcn_s_wait_tensorcnt(0);  // TDM done before barrier
    __syncthreads();

    // --- B fragments: 2x b128 per lane, no packing moves -------------------
    FragU fb0, fb1;
    {
      const __bf16* bp0 = &Xsf[wn * 32 + l16][hl * 8];
      const __bf16* bp1 = &Xsf[wn * 32 + 16 + l16][hl * 8];
      fb0.q[0] = *(const uint4*)bp0; fb0.q[1] = *(const uint4*)(bp0 + 16);
      fb1.q[0] = *(const uint4*)bp1; fb1.q[1] = *(const uint4*)(bp1 + 16);
    }
    // --- A fragments + WMMA -------------------------------------------------
    #pragma unroll
    for (int mm = 0; mm < 4; ++mm) {
      FragU fa;
      const __bf16* ap = &As[wm * 64 + mm * 16 + l16][hl * 8];
      fa.q[0] = *(const uint4*)ap;          // K =      hl*8 .. +7
      fa.q[1] = *(const uint4*)(ap + 16);   // K = 16 + hl*8 .. +7
      acc[mm][0] = __builtin_amdgcn_wmma_f32_16x16x32_bf16(false, fa.v, false, fb0.v,
                                                           (short)0, acc[mm][0], false, false);
      acc[mm][1] = __builtin_amdgcn_wmma_f32_16x16x32_bf16(false, fa.v, false, fb1.v,
                                                           (short)0, acc[mm][1], false, false);
    }
    // --- Fused LoRA t = a_in^T @ x (wave-uniform branch, B-frags reused) ----
    if (wm == 0) {
      FragU fi;
      const __bf16* ip = &Ain[l16][kk + hl * 8];
      fi.q[0] = *(const uint4*)ip;
      fi.q[1] = *(const uint4*)(ip + 16);
      tacc0 = __builtin_amdgcn_wmma_f32_16x16x32_bf16(false, fi.v, false, fb0.v,
                                                      (short)0, tacc0, false, false);
      tacc1 = __builtin_amdgcn_wmma_f32_16x16x32_bf16(false, fi.v, false, fb1.v,
                                                      (short)0, tacc1, false, false);
    }
    __syncthreads();
  }

  // Publish t tile to LDS (C layout: VGPR j -> M=j+8*hl, N=l16)
  if (wm == 0) {
    #pragma unroll
    for (int j = 0; j < 8; ++j) {
      Tld[j + 8 * hl][wn * 32 + l16]      = tacc0[j];
      Tld[j + 8 * hl][wn * 32 + 16 + l16] = tacc1[j];
    }
  }
  __syncthreads();

  // Epilogue: out = acc + bias + sum_r a_out[r][o] * t[r][n]
  #pragma unroll
  for (int mm = 0; mm < 4; ++mm)
    #pragma unroll
    for (int nn = 0; nn < 2; ++nn)
      #pragma unroll
      for (int j = 0; j < 8; ++j) {
        const int ol = wm * 64 + mm * 16 + j + 8 * hl;
        const int nl = wn * 32 + nn * 16 + l16;
        float v = acc[mm][nn][j] + Bld[ol];
        #pragma unroll
        for (int r = 0; r < 8; ++r) v += Aout[r][ol] * Tld[r][nl];
        out[((size_t)b * kCOUT + m0 + ol) * kL + n0 + nl] = v;
      }
}

// ---------------------------------------------------------------------------
extern "C" void kernel_launch(void* const* d_in, const int* in_sizes, int n_in,
                              void* d_out, int out_size, void* d_ws, size_t ws_size,
                              hipStream_t stream) {
  (void)in_sizes; (void)n_in; (void)out_size; (void)ws_size;
  const float* x      = (const float*)d_in[0];
  const float* g      = (const float*)d_in[1];
  const float* main_v = (const float*)d_in[2];
  const float* main_g = (const float*)d_in[3];
  const float* main_b = (const float*)d_in[4];
  const float* ain_v  = (const float*)d_in[5];
  const float* ain_g  = (const float*)d_in[6];
  const float* ain_b  = (const float*)d_in[7];
  const float* aout_v = (const float*)d_in[8];
  const float* aout_g = (const float*)d_in[9];
  const float* aout_b = (const float*)d_in[10];
  float* out = (float*)d_out;

  char* ws = (char*)d_ws;
  __bf16* Wbf   = (__bf16*)(ws);                       // 512*512*2   = 524288 B
  __bf16* ainT  = (__bf16*)(ws + 524288);              // 8*16*512*2  = 131072 B
  float*  aoutW = (float*) (ws + 524288 + 131072);     // 8*8*512*4   = 131072 B

  prep_main_w  <<<dim3(64),   dim3(256), 0, stream>>>(main_v, main_g, Wbf);
  prep_adapters<<<dim3(1024), dim3(256), 0, stream>>>(g, ain_v, ain_g, ain_b,
                                                      aout_v, aout_g, aout_b, ainT, aoutW);
  zero_pad_ain <<<dim3(128),  dim3(256), 0, stream>>>(ainT);
  lora_gemm    <<<dim3(kL/128, kCOUT/128, kB), dim3(256), 0, stream>>>(
      x, Wbf, main_b, ainT, aoutW, out);
}
